// SaProtEpitopeCompareNN_10471130268322
// MI455X (gfx1250) — matrix operand order repaired and checked by reading, not compiled, not run on hardware
//
#include <hip/hip_runtime.h>
#include <stdint.h>
#include <stddef.h>

// ---------------- problem constants (from reference) ----------------
#define B_    2
#define LAG   512
#define LEPI  256
#define C_    446
#define CP    448            // C padded to multiple of 32
#define D_    64
#define KSTEPS (CP/32)       // 14
#define ITILES 4             // 64 i-rows per workgroup (4 x 16)
#define NWAVES 4             // 4 j's per workgroup, one per wave
#define LN_EPS 1e-5f

// ---------------- LDS layout (bytes) ----------------
#define AHI_OFF 0
#define A_BYTES (ITILES*KSTEPS*32*32)          // 57344
#define ALO_OFF (AHI_OFF + A_BYTES)            // 57344
#define W1_OFF  (ALO_OFF + A_BYTES)            // 114688
#define W1_BYTES (D_*CP*4)                     // 114688
#define EPI_OFF (W1_OFF + W1_BYTES)            // 229376
#define EPI_BYTES (NWAVES*CP*4)                // 7168
#define SMEM_BYTES (EPI_OFF + EPI_BYTES)       // 236544 (< 320KB WGP LDS)

typedef __attribute__((ext_vector_type(16))) __bf16 v16bf;
typedef __attribute__((ext_vector_type(8)))  float  v8f;

union BFragU { uint32_t u[8]; v16bf v; };

// round-to-nearest-even f32 -> bf16 bits (staging path only)
__device__ __forceinline__ uint32_t f2bf(float x) {
  uint32_t u = __float_as_uint(x);
  return (u + 0x7FFFu + ((u >> 16) & 1u)) >> 16;
}
__device__ __forceinline__ void split_bf(float x, uint32_t& h, uint32_t& l) {
  h = f2bf(x);
  float hf = __uint_as_float(h << 16);
  l = f2bf(x - hf);
}

// pack the high 16 bits (= truncated bf16) of two f32 bit-patterns: one v_perm_b32
__device__ __forceinline__ uint32_t pack_hi16(uint32_t u0, uint32_t u1) {
  return __builtin_amdgcn_perm(u1, u0, 0x07060302u);
}

__global__ __launch_bounds__(128, 1)
void sapep_fused_wmma_kernel(const float* __restrict__ ag,
                             const float* __restrict__ epi,
                             const float* __restrict__ W1,
                             const float* __restrict__ b1,
                             const float* __restrict__ gamma,
                             const float* __restrict__ beta,
                             const float* __restrict__ W2,
                             const float* __restrict__ b2,
                             float* __restrict__ out)
{
  extern __shared__ char smem[];
  uint32_t* AHI  = (uint32_t*)(smem + AHI_OFF);
  uint32_t* ALO  = (uint32_t*)(smem + ALO_OFF);
  float*    W1s  = (float*)(smem + W1_OFF);
  float*    EPIs = (float*)(smem + EPI_OFF);

  const int tid    = threadIdx.x;
  const int lane   = tid & 31;
  const int wave   = tid >> 5;
  const int n      = lane & 15;   // N-index inside 16-wide tile (== o_local)
  const int laneHi = lane >> 4;

  const int b  = blockIdx.z;
  const int i0 = blockIdx.y * (16 * ITILES);
  const int j0 = blockIdx.x * NWAVES;

  // ---- stage W1 (f32, zero-padded to CP) ----
  for (int p = tid; p < D_ * CP; p += 128) {
    int o = p / CP, c = p - o * CP;
    W1s[p] = (c < C_) ? W1[o * C_ + c] : 0.f;
  }
  // ---- stage the epitope rows (f32, zero-padded) ----
  for (int p = tid; p < NWAVES * CP; p += 128) {
    int w = p / CP, c = p - w * CP;
    EPIs[p] = (c < C_) ? epi[((size_t)(b * LEPI + j0 + w)) * C_ + c] : 0.f;
  }
  // ---- stage ag, split into bf16 hi/lo, PRE-SWIZZLED into WMMA A-layout ----
  for (int p = tid; p < ITILES * KSTEPS * 256; p += 128) {
    int blk = p >> 8;            // it*KSTEPS + ks
    int r   = p & 255;
    int ls  = r >> 3;            // lane slot 0..31
    int pe  = r & 7;             // element pair -> e = 2*pe
    int it  = blk / KSTEPS, ks = blk - it * KSTEPS;
    int m   = ls & 15, hi = ls >> 4;
    int e   = pe * 2;
    int k0  = e + 16 * hi;
    int c   = ks * 32 + k0;
    const float* src = ag + ((size_t)(b * LAG + i0 + it * 16 + m)) * C_;
    float x0 = (c     < C_) ? src[c]     : 0.f;
    float x1 = (c + 1 < C_) ? src[c + 1] : 0.f;
    uint32_t h0, l0, h1, l1;
    split_bf(x0, h0, l0);
    split_bf(x1, h1, l1);
    AHI[p] = h0 | (h1 << 16);
    ALO[p] = l0 | (l1 << 16);
  }
  __syncthreads();

  // ---- main K loop: D = ag x (epi*W1), split-bf16 3-term WMMA ----
  v8f acc[ITILES][4];
  const v8f vzero = {0.f, 0.f, 0.f, 0.f, 0.f, 0.f, 0.f, 0.f};
#pragma unroll
  for (int it = 0; it < ITILES; ++it)
#pragma unroll
    for (int ot = 0; ot < 4; ++ot) acc[it][ot] = vzero;

  const float* epiRow = EPIs + wave * CP;

  for (int ks = 0; ks < KSTEPS; ++ks) {
    // A-fragment hi/lo pairs for this k-step (2x ds_load_b128 each)
    v16bf aHi[ITILES], aLo[ITILES];
#pragma unroll
    for (int it = 0; it < ITILES; ++it) {
      int off = ((it * KSTEPS + ks) * 32 + lane) * 32;
      aHi[it] = *(const v16bf*)(smem + AHI_OFF + off);
      aLo[it] = *(const v16bf*)(smem + ALO_OFF + off);
    }
    // epi values for this k-step (independent of o-tile): 16 consecutive c
    const int cb = ks * 32 + laneHi * 16;
    float4 ecache[4];
    {
      const float4* ep = (const float4*)(epiRow + cb);
#pragma unroll
      for (int q = 0; q < 4; ++q) ecache[q] = ep[q];
    }
#pragma unroll
    for (int ot = 0; ot < 4; ++ot) {
      // Build B fragment (32x16 bf16): lane n holds column o = ot*16+n,
      // K = e + 16*laneHi. Truncation split: hi = top16(x) (1 v_perm per
      // pair), lo = x - hi exact in f32, lo truncated likewise.
      const int o = ot * 16 + n;
      const float4* wp = (const float4*)(W1s + o * CP + cb);
      BFragU bhi, blo;
#pragma unroll
      for (int q = 0; q < 4; ++q) {
        float4 w4 = wp[q];
        float4 e4 = ecache[q];
        float p0 = e4.x * w4.x, p1 = e4.y * w4.y;
        float p2 = e4.z * w4.z, p3 = e4.w * w4.w;
        uint32_t u0 = __float_as_uint(p0), u1 = __float_as_uint(p1);
        uint32_t u2 = __float_as_uint(p2), u3 = __float_as_uint(p3);
        bhi.u[q*2+0] = pack_hi16(u0, u1);
        bhi.u[q*2+1] = pack_hi16(u2, u3);
        float l0 = p0 - __uint_as_float(u0 & 0xFFFF0000u);
        float l1 = p1 - __uint_as_float(u1 & 0xFFFF0000u);
        float l2 = p2 - __uint_as_float(u2 & 0xFFFF0000u);
        float l3 = p3 - __uint_as_float(u3 & 0xFFFF0000u);
        blo.u[q*2+0] = pack_hi16(__float_as_uint(l0), __float_as_uint(l1));
        blo.u[q*2+1] = pack_hi16(__float_as_uint(l2), __float_as_uint(l3));
      }
      // 3-term split-precision accumulate (hi*hi + hi*lo + lo*hi), f32 acc
#pragma unroll
      for (int it = 0; it < ITILES; ++it) {
        acc[it][ot] = __builtin_amdgcn_wmma_f32_16x16x32_bf16(
            false, aHi[it], false, bhi.v, (short)0, acc[it][ot], false, false);
        acc[it][ot] = __builtin_amdgcn_wmma_f32_16x16x32_bf16(
            false, aHi[it], false, blo.v, (short)0, acc[it][ot], false, false);
        acc[it][ot] = __builtin_amdgcn_wmma_f32_16x16x32_bf16(
            false, aLo[it], false, bhi.v, (short)0, acc[it][ot], false, false);
      }
    }
  }

  // ---- epilogue: +b1, LayerNorm(64) over o, ReLU, dot W2, +b2, store ----
  // D layout: VGPR v, lanes 0-15: (M=v,   N=lane)    -> i_local=v,   o=ot*16+n
  //                   lanes 16-31:(M=v+8, N=lane-16) -> i_local=v+8, o=ot*16+n
  float b1v[4], gv[4], bv[4], w2v[4];
#pragma unroll
  for (int ot = 0; ot < 4; ++ot) {
    int o = ot * 16 + n;
    b1v[ot] = b1[o]; gv[ot] = gamma[o]; bv[ot] = beta[o]; w2v[ot] = W2[o];
  }
  const float b2v = b2[0];
  const int j = j0 + wave;

#pragma unroll
  for (int it = 0; it < ITILES; ++it) {
#pragma unroll
    for (int v = 0; v < 8; ++v) {
      float zo[4];
      float s = 0.f, sq = 0.f;
#pragma unroll
      for (int ot = 0; ot < 4; ++ot) {
        zo[ot] = acc[it][ot][v] + b1v[ot];
        s  += zo[ot];
        sq += zo[ot] * zo[ot];
      }
      // reduce across the 16-lane half (o dimension)
#pragma unroll
      for (int msk = 1; msk <= 8; msk <<= 1) {
        s  += __shfl_xor(s,  msk, 32);
        sq += __shfl_xor(sq, msk, 32);
      }
      float mean = s * (1.f / 64.f);
      float var  = sq * (1.f / 64.f) - mean * mean;
      float rstd = rsqrtf(var + LN_EPS);
      float ov = 0.f;
#pragma unroll
      for (int ot = 0; ot < 4; ++ot) {
        float y = (zo[ot] - mean) * rstd * gv[ot] + bv[ot];
        y = fmaxf(y, 0.f);
        ov += y * w2v[ot];
      }
#pragma unroll
      for (int msk = 1; msk <= 8; msk <<= 1) ov += __shfl_xor(ov, msk, 32);
      if (n == 0) {
        int ig = i0 + it * 16 + v + 8 * laneHi;
        out[((size_t)(b * LAG + ig)) * LEPI + j] = ov + b2v;
      }
    }
  }
}

extern "C" void kernel_launch(void* const* d_in, const int* in_sizes, int n_in,
                              void* d_out, int out_size, void* d_ws, size_t ws_size,
                              hipStream_t stream) {
  (void)in_sizes; (void)n_in; (void)out_size; (void)d_ws; (void)ws_size;
  const float* ag    = (const float*)d_in[0];
  const float* epi   = (const float*)d_in[1];
  const float* W1    = (const float*)d_in[2];
  const float* b1    = (const float*)d_in[3];
  const float* gamma = (const float*)d_in[4];
  const float* beta  = (const float*)d_in[5];
  const float* W2    = (const float*)d_in[6];
  const float* b2    = (const float*)d_in[7];
  float* out = (float*)d_out;

  dim3 grid(LEPI / NWAVES, LAG / (16 * ITILES), B_);  // (64, 8, 2) = 1024 WGs
  dim3 block(128);                                    // 4 waves (wave32)
  sapep_fused_wmma_kernel<<<grid, block, SMEM_BYTES, stream>>>(
      ag, epi, W1, b1, gamma, beta, W2, b2, out);
}
